// MixedHead_58935541235897
// MI455X (gfx1250) — matrix operand order repaired and checked by reading, not compiled, not run on hardware
//
#include <hip/hip_runtime.h>
#include <hip/hip_bf16.h>

typedef _Float16 f16;
typedef __attribute__((ext_vector_type(16))) _Float16 v16h;
typedef __attribute__((ext_vector_type(8)))  _Float16 v8h;
typedef __attribute__((ext_vector_type(4)))  _Float16 v4h;
typedef __attribute__((ext_vector_type(8)))  float    v8f;

// ---- problem constants ----
static constexpr int  NB      = 64;
static constexpr int  NT      = 256;
static constexpr long NITEMS  = (long)NB * NT;      // 16384
static constexpr int  DIM     = 512;
static constexpr int  STOCH   = 1024;
static constexpr int  DETER   = 4096;
static constexpr int  H3      = 1536;               // 3*H*HD
static constexpr int  FFD     = 1024;               // 2*D
static constexpr int  NLAYER  = 2;
static constexpr int  OUTD    = 255;
static constexpr float ATT_SCALE = 0.125f;          // 64^-0.5
static constexpr float LN_EPS    = 1e-5f;

// ======================================================================
// wave32 reduction
// ======================================================================
__device__ __forceinline__ float wave_sum(float v) {
#pragma unroll
  for (int off = 16; off > 0; off >>= 1) v += __shfl_xor(v, off, 32);
  return v;
}

// ======================================================================
// f32 -> f16 conversion (weights)
// ======================================================================
__global__ void cvt_f16_kernel(const float* __restrict__ s, f16* __restrict__ d, long n) {
  long i = (long)blockIdx.x * blockDim.x + threadIdx.x;
  if (i < n) d[i] = (f16)s[i];
}

// ======================================================================
// embedding gather: q2[n, d] = emb[tg[n], d]
// ======================================================================
__global__ void gather_emb_kernel(const int* __restrict__ tg, const float* __restrict__ emb,
                                  float* __restrict__ q2) {
  long idx = (long)blockIdx.x * blockDim.x + threadIdx.x;  // N*512 threads
  long n = idx >> 9;
  int  d = (int)(idx & 511);
  q2[idx] = emb[(long)tg[n] * DIM + d];
}

// ======================================================================
// LayerNorm over D=512, one wave per row
// ======================================================================
__global__ __launch_bounds__(256)
void ln_kernel(const float* __restrict__ x, float* __restrict__ y,
               const float* __restrict__ g, const float* __restrict__ b, long nrows) {
  long row = (long)blockIdx.x * 8 + (threadIdx.x >> 5);
  if (row >= nrows) return;
  int lane = threadIdx.x & 31;
  const float* xr = x + row * DIM;

  float4 vbuf[4];
  float s = 0.f, sq = 0.f;
#pragma unroll
  for (int i = 0; i < 4; ++i) {
    int c = (i * 32 + lane) * 4;
    float4 v = *(const float4*)(xr + c);
    vbuf[i] = v;
    s  += v.x + v.y + v.z + v.w;
    sq += v.x * v.x + v.y * v.y + v.z * v.z + v.w * v.w;
  }
  s  = wave_sum(s);
  sq = wave_sum(sq);
  float mean = s * (1.0f / DIM);
  float var  = sq * (1.0f / DIM) - mean * mean;
  float inv  = rsqrtf(var + LN_EPS);

  float* yr = y + row * DIM;
#pragma unroll
  for (int i = 0; i < 4; ++i) {
    int c = (i * 32 + lane) * 4;
    float4 gg = *(const float4*)(g + c);
    float4 bb = *(const float4*)(b + c);
    float4 v = vbuf[i], o;
    o.x = (v.x - mean) * inv * gg.x + bb.x;
    o.y = (v.y - mean) * inv * gg.y + bb.y;
    o.z = (v.z - mean) * inv * gg.z + bb.z;
    o.w = (v.w - mean) * inv * gg.w + bb.w;
    *(float4*)(yr + c) = o;
  }
}

// ======================================================================
// Attention: one wave per item. seq=2 self tokens, 1 cross query,
// 8 heads x 64 dims. log-softmax attention (as in reference).
// ======================================================================
__device__ __forceinline__ void lsm2(float a, float b, float& oa, float& ob) {
  float m = fmaxf(a, b);
  float l = m + logf(expf(a - m) + expf(b - m));
  oa = a - l; ob = b - l;
}

__global__ __launch_bounds__(256)
void attn_kernel(const float* __restrict__ qkv, const float* __restrict__ qn,
                 float* __restrict__ att, float* __restrict__ q2) {
  long n = (long)blockIdx.x * 8 + (threadIdx.x >> 5);
  int lane = threadIdx.x & 31;
  const float* base = qkv + n * 2 * H3;

#pragma unroll 1
  for (int h = 0; h < 8; ++h) {
    int da = h * 64 + lane;
    int db = da + 32;
    float q0a = base[da],            q0b = base[db];
    float q1a = base[H3 + da],       q1b = base[H3 + db];
    float k0a = base[512 + da],      k0b = base[512 + db];
    float k1a = base[H3 + 512 + da], k1b = base[H3 + 512 + db];
    float qca = qn[n * DIM + da],    qcb = qn[n * DIM + db];

    float s00 = q0a * k0a + q0b * k0b;
    float s01 = q0a * k1a + q0b * k1b;
    float s10 = q1a * k0a + q1b * k0b;
    float s11 = q1a * k1a + q1b * k1b;
    float sc0 = qca * k0a + qcb * k0b;
    float sc1 = qca * k1a + qcb * k1b;

    s00 = wave_sum(s00) * ATT_SCALE;
    s01 = wave_sum(s01) * ATT_SCALE;
    s10 = wave_sum(s10) * ATT_SCALE;
    s11 = wave_sum(s11) * ATT_SCALE;
    sc0 = wave_sum(sc0) * ATT_SCALE;
    sc1 = wave_sum(sc1) * ATT_SCALE;

    float a00, a01, a10, a11, c0, c1;
    lsm2(s00, s01, a00, a01);
    lsm2(s10, s11, a10, a11);
    lsm2(sc0, sc1, c0, c1);

    float v0a = base[1024 + da],      v0b = base[1024 + db];
    float v1a = base[H3 + 1024 + da], v1b = base[H3 + 1024 + db];

    att[(n * 2) * DIM + da]     = a00 * v0a + a01 * v1a;
    att[(n * 2) * DIM + db]     = a00 * v0b + a01 * v1b;
    att[(n * 2 + 1) * DIM + da] = a10 * v0a + a11 * v1a;
    att[(n * 2 + 1) * DIM + db] = a10 * v0b + a11 * v1b;
    q2[n * DIM + da] = c0 * v0a + c1 * v1a + qca;
    q2[n * DIM + db] = c0 * v0b + c1 * v1b + qcb;
  }
}

// ======================================================================
// Software-pipelined WMMA GEMM:
//   C[r, col_off + c] = act( A[r,:K] . W[c,:K] + bias[c] ) (+res)
//   A: f32 [rows,K] row-major; W: f16 [Nc,K] row-major.
// Block = 128x128 tile, BK=32, 256 threads = 8 waves (4Mx2N), wave = 32x64
// via 2x4 v_wmma_f32_16x16x32_f16, f32 accumulate.
// Pipeline: global loads for tile k+1 are issued BEFORE computing tile k,
// conversion + LDS store happen after the compute barrier, so the
// s_wait_loadcnt is hidden behind the 8 WMMAs + fragment ds_loads.
// ======================================================================
__global__ __launch_bounds__(256)
void gemm_wmma_f16(const float* __restrict__ A, const f16* __restrict__ W,
                   const float* __restrict__ bias, const float* __restrict__ res,
                   float* __restrict__ C,
                   int K, int Nc, int ldc, int col_off, int rld, int rcol_off,
                   int actGelu) {
  __shared__ f16 sA[128 * 40];   // [row][k], padded stride 40 halfs
  __shared__ f16 sB[128 * 40];   // [col][k]

  const int tid  = threadIdx.x;
  const int lane = tid & 31;
  const int wid  = tid >> 5;
  const int wm   = wid & 3;      // 0..3  -> 32-row slice
  const int wn   = wid >> 2;     // 0..1  -> 64-col slice
  const int l15  = lane & 15;
  const int hi   = lane >> 4;    // half-wave select
  const long rowBase = (long)blockIdx.x * 128;
  const int  colBase = blockIdx.y * 128;

  // per-thread staging coordinates (same for every K tile)
  int rA[4], kA[4], nB[4], kB[4];
  bool okB[4];
#pragma unroll
  for (int i = 0; i < 4; ++i) {
    int idx4 = i * 256 + tid;
    rA[i] = idx4 >> 3;  kA[i] = (idx4 & 7) << 2;
    nB[i] = idx4 >> 3;  kB[i] = (idx4 & 7) << 2;
    okB[i] = (colBase + nB[i]) < Nc;
  }

  float4 ra[4];   // staged A (f32)
  v4h    rb[4];   // staged B (f16)

  auto load_tiles = [&](int kb) {
#pragma unroll
    for (int i = 0; i < 4; ++i)
      ra[i] = *(const float4*)(A + (rowBase + rA[i]) * (long)K + kb + kA[i]);
#pragma unroll
    for (int i = 0; i < 4; ++i) {
      if (okB[i]) {
        rb[i] = *(const v4h*)(W + (long)(colBase + nB[i]) * K + kb + kB[i]);
      } else {
        rb[i][0] = (f16)0.f; rb[i][1] = (f16)0.f;
        rb[i][2] = (f16)0.f; rb[i][3] = (f16)0.f;
      }
    }
  };
  auto store_tiles = [&]() {
#pragma unroll
    for (int i = 0; i < 4; ++i) {
      v4h h;
      h[0] = (f16)ra[i].x; h[1] = (f16)ra[i].y;
      h[2] = (f16)ra[i].z; h[3] = (f16)ra[i].w;
      *(v4h*)(&sA[rA[i] * 40 + kA[i]]) = h;
    }
#pragma unroll
    for (int i = 0; i < 4; ++i)
      *(v4h*)(&sB[nB[i] * 40 + kB[i]]) = rb[i];
  };

  v8f acc[2][4];
#pragma unroll
  for (int i = 0; i < 2; ++i)
#pragma unroll
    for (int j = 0; j < 4; ++j)
#pragma unroll
      for (int e = 0; e < 8; ++e) acc[i][j][e] = 0.0f;

  // prologue: stage tile 0
  load_tiles(0);
  store_tiles();
  __syncthreads();

  for (int kb = 0; kb < K; kb += 32) {
    const bool last = (kb + 32 >= K);
    // issue next tile's global loads (no dependent use yet -> no wait here)
    if (!last) {
      load_tiles(kb + 32);
      if (kb + 64 < K)
        __builtin_prefetch(A + (rowBase + rA[0]) * (long)K + kb + 64 + kA[0], 0, 1);
    }

    // ---- compute current tile from LDS ----
    // A 16x32: lane<16 -> M=l15, K 0..7 & 16..23 ; lane>=16 -> K 8..15 & 24..31
    v16h af[2];
#pragma unroll
    for (int fm = 0; fm < 2; ++fm) {
      const f16* p = &sA[(wm * 32 + fm * 16 + l15) * 40 + hi * 8];
      v8h x0 = *(const v8h*)(p);
      v8h x1 = *(const v8h*)(p + 16);
#pragma unroll
      for (int e = 0; e < 8; ++e) { af[fm][e] = x0[e]; af[fm][8 + e] = x1[e]; }
    }
    // B 32x16: lane<16 -> N=l15, K 0..7 & 16..23 ; lane>=16 -> K 8..15 & 24..31
    v16h bf[4];
#pragma unroll
    for (int fn = 0; fn < 4; ++fn) {
      const f16* p = &sB[(wn * 64 + fn * 16 + l15) * 40 + hi * 8];
      v8h x0 = *(const v8h*)(p);
      v8h x1 = *(const v8h*)(p + 16);
#pragma unroll
      for (int e = 0; e < 8; ++e) { bf[fn][e] = x0[e]; bf[fn][8 + e] = x1[e]; }
    }

#pragma unroll
    for (int fm = 0; fm < 2; ++fm)
#pragma unroll
      for (int fn = 0; fn < 4; ++fn)
        acc[fm][fn] = __builtin_amdgcn_wmma_f32_16x16x32_f16(
            false, af[fm], false, bf[fn], (short)0, acc[fm][fn], false, false);

    // ---- rotate pipeline ----
    __syncthreads();              // all waves done reading LDS for this tile
    if (!last) {
      store_tiles();              // wait for in-flight loads happens here
      __syncthreads();
    }
  }

  // ---- epilogue: bias, optional GELU, optional residual, store ----
  // C/D layout: lane<16 -> N=l15, M=e ; lane>=16 -> N=l15, M=e+8
#pragma unroll
  for (int fm = 0; fm < 2; ++fm)
#pragma unroll
    for (int fn = 0; fn < 4; ++fn) {
      int c = colBase + wn * 64 + fn * 16 + l15;
      if (c < Nc) {
        float bv = bias ? bias[c] : 0.0f;
        long r0 = rowBase + wm * 32 + fm * 16 + hi * 8;
#pragma unroll
        for (int e = 0; e < 8; ++e) {
          long r = r0 + e;
          float val = acc[fm][fn][e] + bv;
          if (actGelu) val = 0.5f * val * (1.0f + erff(val * 0.70710678118654752f));
          if (res) val += res[r * (long)rld + rcol_off + c];
          C[r * (long)ldc + col_off + c] = val;
        }
      }
    }
}

// ======================================================================
// host
// ======================================================================
extern "C" void kernel_launch(void* const* d_in, const int* in_sizes, int n_in,
                              void* d_out, int out_size, void* d_ws, size_t ws_size,
                              hipStream_t stream) {
  (void)in_sizes; (void)n_in; (void)out_size; (void)ws_size;
  const float* stoch = (const float*)d_in[0];
  const float* deter = (const float*)d_in[1];
  const int*   tg    = (const int*)d_in[2];
  const float* emb   = (const float*)d_in[3];
  const float* Ws    = (const float*)d_in[4];
  const float* bs    = (const float*)d_in[5];
  const float* Wd    = (const float*)d_in[6];
  const float* bd    = (const float*)d_in[7];
  const float* ln1g  = (const float*)d_in[8];
  const float* ln1b  = (const float*)d_in[9];
  const float* Wqkv  = (const float*)d_in[10];
  const float* Wout  = (const float*)d_in[11];
  const float* bout  = (const float*)d_in[12];
  const float* ln2g  = (const float*)d_in[13];
  const float* ln2b  = (const float*)d_in[14];
  const float* W1    = (const float*)d_in[15];
  const float* b1    = (const float*)d_in[16];
  const float* W2    = (const float*)d_in[17];
  const float* b2    = (const float*)d_in[18];
  const float* W1b   = (const float*)d_in[19];
  const float* b1b   = (const float*)d_in[20];
  const float* W2b   = (const float*)d_in[21];
  const float* b2b   = (const float*)d_in[22];
  const float* Wm    = (const float*)d_in[23];
  const float* bm    = (const float*)d_in[24];
  float* out = (float*)d_out;

  // ---- workspace carve-up ----
  char* ws = (char*)d_ws;
  size_t off = 0;
  auto take = [&](size_t bytes) {
    char* p = ws + off;
    off += (bytes + 255) & ~(size_t)255;
    return (void*)p;
  };
  const long N2 = NITEMS * 2;
  float* X   = (float*)take((size_t)N2 * DIM * 4);       // [N,2,D]
  float* Q   = (float*)take((size_t)NITEMS * DIM * 4);   // [N,D]
  float* XN  = (float*)take((size_t)N2 * DIM * 4);
  float* QN  = (float*)take((size_t)NITEMS * DIM * 4);
  float* QKV = (float*)take((size_t)N2 * H3 * 4);        // also self FF hidden
  float* ATT = (float*)take((size_t)N2 * DIM * 4);       // also cross FF hidden
  f16* Ws_h   = (f16*)take((size_t)DIM * STOCH * 2);
  f16* Wd_h   = (f16*)take((size_t)DIM * DETER * 2);
  f16* Wqkv_h = (f16*)take((size_t)NLAYER * H3 * DIM * 2);
  f16* Wout_h = (f16*)take((size_t)NLAYER * DIM * DIM * 2);
  f16* W1_h   = (f16*)take((size_t)NLAYER * FFD * DIM * 2);
  f16* W2_h   = (f16*)take((size_t)NLAYER * DIM * FFD * 2);
  f16* W1b_h  = (f16*)take((size_t)NLAYER * FFD * DIM * 2);
  f16* W2b_h  = (f16*)take((size_t)NLAYER * DIM * FFD * 2);
  f16* Wm_h   = (f16*)take((size_t)OUTD * DIM * 2);

  auto cvt = [&](const float* s, f16* d, long n) {
    cvt_f16_kernel<<<(unsigned)((n + 255) / 256), 256, 0, stream>>>(s, d, n);
  };
  cvt(Ws,   Ws_h,   (long)DIM * STOCH);
  cvt(Wd,   Wd_h,   (long)DIM * DETER);
  cvt(Wqkv, Wqkv_h, (long)NLAYER * H3 * DIM);
  cvt(Wout, Wout_h, (long)NLAYER * DIM * DIM);
  cvt(W1,   W1_h,   (long)NLAYER * FFD * DIM);
  cvt(W2,   W2_h,   (long)NLAYER * DIM * FFD);
  cvt(W1b,  W1b_h,  (long)NLAYER * FFD * DIM);
  cvt(W2b,  W2b_h,  (long)NLAYER * DIM * FFD);
  cvt(Wm,   Wm_h,   (long)OUTD * DIM);

  auto gemm = [&](const float* A, const f16* W, const float* bias, const float* res,
                  float* C, int K, int Nc, int ldc, int col_off, int rld, int rcoff,
                  int act, long rows) {
    dim3 g((unsigned)(rows / 128), (unsigned)((Nc + 127) / 128));
    gemm_wmma_f16<<<g, 256, 0, stream>>>(A, W, bias, res, C, K, Nc, ldc, col_off,
                                         rld, rcoff, act);
  };
  auto ln = [&](const float* x, float* y, const float* g, const float* b, long rows) {
    ln_kernel<<<(unsigned)((rows + 7) / 8), 256, 0, stream>>>(x, y, g, b, rows);
  };

  // ---- token embeddings ----
  gather_emb_kernel<<<(unsigned)(NITEMS * DIM / 256), 256, 0, stream>>>(tg, emb, Q);
  // x[:,0,:] = stoch @ Ws^T + bs ; x[:,1,:] = deter @ Wd^T + bd
  gemm(stoch, Ws_h, bs, nullptr, X, STOCH, DIM, 2 * DIM, 0,   0, 0, 0, NITEMS);
  gemm(deter, Wd_h, bd, nullptr, X, DETER, DIM, 2 * DIM, DIM, 0, 0, 0, NITEMS);

  // ---- transformer layers ----
  for (int l = 0; l < NLAYER; ++l) {
    const float* g1 = ln1g + l * DIM; const float* bb1 = ln1b + l * DIM;
    const float* g2 = ln2g + l * DIM; const float* bb2 = ln2b + l * DIM;

    // PreNorm + attention
    ln(X, XN, g1, bb1, N2);
    ln(Q, QN, g1, bb1, NITEMS);
    gemm(XN, Wqkv_h + (size_t)l * H3 * DIM, nullptr, nullptr,
         QKV, DIM, H3, H3, 0, 0, 0, 0, N2);
    attn_kernel<<<(unsigned)(NITEMS / 8), 256, 0, stream>>>(QKV, QN, ATT, Q);
    // x = att @ Wout^T + bout + xn
    gemm(ATT, Wout_h + (size_t)l * DIM * DIM, bout + l * DIM, XN,
         X, DIM, DIM, DIM, 0, DIM, 0, 0, N2);

    // PreNorm + feedforward (self)
    ln(X, XN, g2, bb2, N2);
    gemm(XN, W1_h + (size_t)l * FFD * DIM, b1 + l * FFD, nullptr,
         QKV, DIM, FFD, FFD, 0, 0, 0, /*gelu=*/1, N2);
    gemm(QKV, W2_h + (size_t)l * DIM * FFD, b2 + l * DIM, XN,
         X, FFD, DIM, DIM, 0, DIM, 0, 0, N2);

    // PreNorm + feedforward (cross)
    ln(Q, QN, g2, bb2, NITEMS);
    gemm(QN, W1b_h + (size_t)l * FFD * DIM, b1b + l * FFD, nullptr,
         ATT, DIM, FFD, FFD, 0, 0, 0, /*gelu=*/1, NITEMS);
    gemm(ATT, W2b_h + (size_t)l * DIM * FFD, b2b + l * DIM, QN,
         Q, FFD, DIM, DIM, 0, DIM, 0, 0, NITEMS);
  }

  // ---- head: out = q2 @ Wm^T + bm ----
  gemm(Q, Wm_h, bm, nullptr, out, DIM, OUTD, OUTD, 0, 0, 0, 0, NITEMS);
}
